// StatefulBlock_85899346617
// MI455X (gfx1250) — matrix operand order repaired and checked by reading, not compiled, not run on hardware
//
#include <hip/hip_runtime.h>
#include <hip/hip_bf16.h>

#define EPSF 1.1920929e-07f

typedef __attribute__((ext_vector_type(16))) __bf16          v16bf;
typedef __attribute__((ext_vector_type(8)))  float           v8f;
typedef __attribute__((ext_vector_type(8)))  unsigned short  ushort8;
typedef __attribute__((ext_vector_type(16))) unsigned short  ushort16;

union Frag {
    ushort16 u;
    v16bf    b;
    ushort8  h[2];
};

__device__ __forceinline__ unsigned short f2b(float f) {
    unsigned int u = __float_as_uint(f);
    unsigned int r = (u + 0x7FFFu + ((u >> 16) & 1u)) >> 16;
    return (unsigned short)r;
}
__device__ __forceinline__ float b2f(unsigned short s) {
    return __uint_as_float(((unsigned int)s) << 16);
}

// ---------------------------------------------------------------------------
// FiLM params: state_mod = state @ w_state_mod; fg = gamma*(1+sm[:D]); fb = beta+sm[D:]
// ---------------------------------------------------------------------------
__global__ void film_kernel(const float* __restrict__ state, const float* __restrict__ wsm,
                            const float* __restrict__ gamma, const float* __restrict__ beta,
                            float* __restrict__ fg, float* __restrict__ fb, int SD, int D) {
    int idx = blockIdx.x * blockDim.x + threadIdx.x;     // over B * 2D
    int b = idx / (2 * D);
    int c = idx % (2 * D);
    if (b >= 2) return;
    float s = 0.f;
    for (int k = 0; k < SD; ++k) s += state[b * SD + k] * wsm[k * 2 * D + c];
    if (c < D) fg[b * D + c] = gamma[c] * (1.f + s);
    else       fb[b * D + (c - D)] = beta[c - D] + s;
}

// ---------------------------------------------------------------------------
// RMSNorm + FiLM -> bf16 row-major output
// ---------------------------------------------------------------------------
__global__ __launch_bounds__(256) void rmsnorm_film(
        const float* __restrict__ x, const float* __restrict__ nw,
        const float* __restrict__ fg, const float* __restrict__ fb,
        unsigned short* __restrict__ out, int S, int D) {
    int row = blockIdx.x;           // b*S + s
    int b = row / S;
    const float* xr = x + (size_t)row * D;
    float ss = 0.f;
    for (int d = threadIdx.x; d < D; d += 256) { float v = xr[d]; ss += v * v; }
    for (int m = 16; m >= 1; m >>= 1) ss += __shfl_xor(ss, m, 32);
    __shared__ float red[8];
    if ((threadIdx.x & 31) == 0) red[threadIdx.x >> 5] = ss;
    __syncthreads();
    float tot = 0.f;
    for (int i = 0; i < 8; ++i) tot += red[i];
    float scale = rsqrtf(tot / (float)D + EPSF);
    const float* fgb = fg + (size_t)b * D;
    const float* fbb = fb + (size_t)b * D;
    unsigned short* orow = out + (size_t)row * D;
    for (int d = threadIdx.x; d < D; d += 256)
        orow[d] = f2b(xr[d] * scale * nw[d] * fgb[d] + fbb[d]);
}

// ---------------------------------------------------------------------------
// Pack f32 weight (K x N, row-major, y = x@W) into WMMA B-fragment layout:
// frag (nt, kt): 32 lanes x 16 halfs; lane<16: B[k0+(j&7)+16*(j>>3)][nt*16+lane], lane>=16: same k + 8
// ---------------------------------------------------------------------------
__global__ void pack_b(const float* __restrict__ W, unsigned short* __restrict__ out, int K, int N) {
    int idx = blockIdx.x * 256 + threadIdx.x;
    if (idx >= K * N) return;
    int j    = idx & 15;
    int lane = (idx >> 4) & 31;
    int r    = idx >> 9;
    int ktc  = K >> 5;
    int kt   = r % ktc;
    int nt   = r / ktc;
    int k = kt * 32 + ((lane >= 16) ? 8 : 0) + (j & 7) + ((j >= 8) ? 16 : 0);
    int n = nt * 16 + (lane & 15);
    out[idx] = f2b(W[(size_t)k * N + n]);
}

// ---------------------------------------------------------------------------
// Pack V (from bf16 qkv buffer) into B-fragments: [bh][kb][n4][lane][16]
// ---------------------------------------------------------------------------
__global__ void pack_v(const unsigned short* __restrict__ qkvb, unsigned short* __restrict__ vp,
                       int S, int H, int HD, int total) {
    int idx = blockIdx.x * 256 + threadIdx.x;
    if (idx >= total) return;
    int j    = idx & 15;
    int lane = (idx >> 4) & 31;
    int r    = idx >> 9;
    int n4 = r & 3;  r >>= 2;
    int kb = r & 63; r >>= 6;
    int bh = r;
    int b = bh >> 4, h = bh & 15;
    int key = kb * 32 + ((lane >= 16) ? 8 : 0) + (j & 7) + ((j >= 8) ? 16 : 0);
    int col = n4 * 16 + (lane & 15);
    vp[idx] = qkvb[(size_t)(b * S + key) * (3 * H * HD) + 2 * H * HD + h * HD + col];
}

// ---------------------------------------------------------------------------
// bf16 GEMM via V_WMMA_F32_16X16X32_BF16.
// Block: 256 thr = 8 waves; wave tile 32x64 (2x4 accum frags); block tile 64x256.
// A (M x K bf16 row-major) is double-buffer staged into LDS per K-step using
// gfx1250 async copy (GLOBAL_LOAD_ASYNC_TO_LDS_B128 + s_wait_asynccnt).
// B streams from packed fragments (contiguous 32B/lane).
// Epilogues: 0: bf16 store | 1: f32 = acc+res | 2: bf16 silu(acc) | 3: bf16 gate*acc
// ---------------------------------------------------------------------------
template <int EPI>
__global__ __launch_bounds__(256) void gemm_bf16(
        const unsigned short* __restrict__ A, const unsigned short* __restrict__ Bp,
        const float* __restrict__ resF, const unsigned short* __restrict__ gateB,
        float* __restrict__ outF, unsigned short* __restrict__ outB,
        int M, int N, int K) {
    __shared__ __align__(16) unsigned short atile[2][64 * 32];   // 2 x 4KB

    const int lane = threadIdx.x & 31;
    const int w    = threadIdx.x >> 5;
    const int wm   = w >> 2;
    const int wn   = w & 3;
    const int mblock = blockIdx.y * 64;
    const int mbase  = mblock + wm * 32;
    const int nbase  = blockIdx.x * 256 + wn * 64;
    const int ktc = K >> 5;
    const int hi  = (lane >> 4) & 1;
    const int l15 = lane & 15;

    // cooperative A-staging mapping: 256 thr x 16B = 64x32 bf16 tile
    const int srow   = threadIdx.x >> 2;
    const int schunk = (threadIdx.x & 3) * 8;            // ushort offset within row

    v8f acc[2][4] = {};

    const unsigned short* bp[4];
    for (int ni = 0; ni < 4; ++ni)
        bp[ni] = Bp + ((size_t)((nbase >> 4) + ni) * ktc) * 512 + lane * 16;

    const unsigned short* ag = A + (size_t)(mblock + srow) * K + schunk;

    // issue async copy of A tile for step kt into buffer buf
    auto issue = [&](int buf, int kt) {
        unsigned long long g = (unsigned long long)(ag + kt * 32);
        unsigned int l = (unsigned int)(size_t)(&atile[buf][srow * 32 + schunk]);
        asm volatile("global_load_async_to_lds_b128 %0, %1, off"
                     :: "v"(l), "v"(g) : "memory");
    };

    issue(0, 0);
    int cur = 0;

    for (int kt = 0; kt < ktc; ++kt) {
        asm volatile("s_wait_asynccnt 0" ::: "memory");  // own async writes done
        __syncthreads();                                  // everyone's tile visible
        if (kt + 1 < ktc) issue(cur ^ 1, kt + 1);

        Frag bfr[4];
        for (int ni = 0; ni < 4; ++ni) {
            bfr[ni].u = *(const ushort16*)(bp[ni]);
            if (kt + 1 < ktc) __builtin_prefetch(bp[ni] + 512, 0, 0);
            bp[ni] += 512;
        }

        Frag a0, a1;
        const unsigned short* l0 = &atile[cur][(wm * 32 + l15) * 32 + hi * 8];
        a0.h[0] = *(const ushort8*)(l0);
        a0.h[1] = *(const ushort8*)(l0 + 16);
        const unsigned short* l1 = l0 + 16 * 32;
        a1.h[0] = *(const ushort8*)(l1);
        a1.h[1] = *(const ushort8*)(l1 + 16);

        for (int ni = 0; ni < 4; ++ni) {
            acc[0][ni] = __builtin_amdgcn_wmma_f32_16x16x32_bf16(false, a0.b, false, bfr[ni].b, (short)0, acc[0][ni], false, false);
            acc[1][ni] = __builtin_amdgcn_wmma_f32_16x16x32_bf16(false, a1.b, false, bfr[ni].b, (short)0, acc[1][ni], false, false);
        }
        cur ^= 1;
    }

    for (int mi = 0; mi < 2; ++mi) {
        for (int ni = 0; ni < 4; ++ni) {
            int rbase = mbase + mi * 16 + hi * 8;
            int col   = nbase + ni * 16 + l15;
            for (int r = 0; r < 8; ++r) {
                size_t idx = (size_t)(rbase + r) * N + col;
                float v = acc[mi][ni][r];
                if (EPI == 0) {
                    outB[idx] = f2b(v);
                } else if (EPI == 1) {
                    outF[idx] = v + resF[idx];
                } else if (EPI == 2) {
                    outB[idx] = f2b(v / (1.f + __expf(-v)));   // silu
                } else {
                    outB[idx] = f2b(b2f(gateB[idx]) * v);
                }
            }
        }
    }
}

// ---------------------------------------------------------------------------
// Flash attention, one wave per 16-query tile; causal; online softmax.
// ---------------------------------------------------------------------------
__global__ __launch_bounds__(256) void flash_attn(
        const unsigned short* __restrict__ qkvb, const unsigned short* __restrict__ vp,
        unsigned short* __restrict__ attn, int S, int H, int HD) {
    __shared__ __align__(16) unsigned short plds[8][512];   // per-wave 16x32 bf16 P staging
    const int lane = threadIdx.x & 31;
    const int w    = threadIdx.x >> 5;
    const int waveId = blockIdx.x * 8 + w;
    const int qt = waveId & 127;              // S/16 = 128
    const int bh = waveId >> 7;
    const int b = bh >> 4, h = bh & 15;
    const int qbase = qt * 16;
    const int l15 = lane & 15;
    const int hi  = (lane >> 4) & 1;
    const int D3  = 3 * H * HD;

    Frag aq[2];
    {
        const unsigned short* qp = qkvb + (size_t)(b * S + qbase + l15) * D3 + h * HD + hi * 16;
        aq[0].h[0] = *(const ushort8*)(qp);
        aq[0].h[1] = *(const ushort8*)(qp + 8);
        aq[1].h[0] = *(const ushort8*)(qp + 32);
        aq[1].h[1] = *(const ushort8*)(qp + 40);
    }

    v8f o[4] = {};
    float mvec[8], lvec[8];
    for (int r = 0; r < 8; ++r) { mvec[r] = -3.0e38f; lvec[r] = 0.f; }

    const int nkb = (qbase + 16 + 31) >> 5;
    for (int kb = 0; kb < nkb; ++kb) {
        const int kbase = kb * 32;
        Frag bk[2][2];
        for (int t = 0; t < 2; ++t) {
            const unsigned short* kp =
                qkvb + (size_t)(b * S + kbase + t * 16 + l15) * D3 + H * HD + h * HD + hi * 16;
            bk[t][0].u = *(const ushort16*)(kp);
            bk[t][1].u = *(const ushort16*)(kp + 32);
        }
        v8f z = {};
        v8f sc[2];
        sc[0] = __builtin_amdgcn_wmma_f32_16x16x32_bf16(false, aq[0].b, false, bk[0][0].b, (short)0, z, false, false);
        sc[0] = __builtin_amdgcn_wmma_f32_16x16x32_bf16(false, aq[1].b, false, bk[0][1].b, (short)0, sc[0], false, false);
        sc[1] = __builtin_amdgcn_wmma_f32_16x16x32_bf16(false, aq[0].b, false, bk[1][0].b, (short)0, z, false, false);
        sc[1] = __builtin_amdgcn_wmma_f32_16x16x32_bf16(false, aq[1].b, false, bk[1][1].b, (short)0, sc[1], false, false);

        const bool diag = (kbase + 32 > qbase);
        for (int t = 0; t < 2; ++t)
            for (int r = 0; r < 8; ++r) {
                float v = sc[t][r] * 0.125f;            // 1/sqrt(64)
                if (diag) {
                    int key  = kbase + t * 16 + l15;
                    int qrow = qbase + r + hi * 8;
                    if (key > qrow) v = -1.0e30f;
                }
                sc[t][r] = v;
            }

        float al[8];
        for (int r = 0; r < 8; ++r) {
            float v = fmaxf(sc[0][r], sc[1][r]);
            v = fmaxf(v, __shfl_xor(v, 1, 32));
            v = fmaxf(v, __shfl_xor(v, 2, 32));
            v = fmaxf(v, __shfl_xor(v, 4, 32));
            v = fmaxf(v, __shfl_xor(v, 8, 32));
            float mn = fmaxf(mvec[r], v);
            al[r] = __expf(mvec[r] - mn);
            mvec[r] = mn;
        }
        for (int t = 0; t < 2; ++t)
            for (int r = 0; r < 8; ++r)
                sc[t][r] = __expf(sc[t][r] - mvec[r]);
        for (int r = 0; r < 8; ++r) {
            float rs = sc[0][r] + sc[1][r];
            rs += __shfl_xor(rs, 1, 32);
            rs += __shfl_xor(rs, 2, 32);
            rs += __shfl_xor(rs, 4, 32);
            rs += __shfl_xor(rs, 8, 32);
            lvec[r] = lvec[r] * al[r] + rs;
        }
        for (int n = 0; n < 4; ++n)
            for (int r = 0; r < 8; ++r)
                o[n][r] *= al[r];

        // transpose P (C layout -> A layout) through LDS (wave-private region)
        for (int t = 0; t < 2; ++t)
            for (int r = 0; r < 8; ++r)
                plds[w][(r + hi * 8) * 32 + t * 16 + l15] = f2b(sc[t][r]);
        asm volatile("s_wait_dscnt 0" ::: "memory");
        Frag pa;
        pa.h[0] = *(const ushort8*)&plds[w][l15 * 32 + hi * 8];
        pa.h[1] = *(const ushort8*)&plds[w][l15 * 32 + hi * 8 + 16];

        const unsigned short* vpp = vp + (((size_t)bh * (S / 32) + kb) * 4) * 512 + lane * 16;
        for (int n = 0; n < 4; ++n) {
            Frag bv;
            bv.u = *(const ushort16*)(vpp + n * 512);
            o[n] = __builtin_amdgcn_wmma_f32_16x16x32_bf16(false, pa.b, false, bv.b, (short)0, o[n], false, false);
        }
        asm volatile("s_wait_dscnt 0" ::: "memory");
    }

    for (int n = 0; n < 4; ++n)
        for (int r = 0; r < 8; ++r) {
            int row = qbase + r + hi * 8;
            int col = h * HD + n * 16 + l15;
            attn[(size_t)(b * S + row) * (H * HD) + col] = f2b(o[n][r] / lvec[r]);
        }
}

// ---------------------------------------------------------------------------
// mean over S:  out[b,d] = mean_s x[b,s,d]
// ---------------------------------------------------------------------------
__global__ void colmean(const float* __restrict__ x, float* __restrict__ out, int B, int S, int D) {
    int idx = blockIdx.x * 256 + threadIdx.x;
    if (idx >= B * D) return;
    int b = idx / D, d = idx % D;
    float s = 0.f;
    for (int t = 0; t < S; ++t) s += x[((size_t)b * S + t) * D + d];
    out[idx] = s / (float)S;
}

// ---------------------------------------------------------------------------
// recurrent state update (tiny): one block, 128 threads
// ---------------------------------------------------------------------------
__global__ __launch_bounds__(128) void state_update(
        const float* __restrict__ state, const float* __restrict__ meanx,
        const float* __restrict__ w_h2s, const float* __restrict__ w_sgate,
        float* __restrict__ outState, int D, int SD) {
    __shared__ float ni[2][64];
    int t = threadIdx.x;
    int b = t >> 6, j = t & 63;
    float s = 0.f;
    for (int k = 0; k < D; ++k) s += meanx[b * D + k] * w_h2s[(size_t)k * SD + j];
    ni[b][j] = s;
    __syncthreads();
    float g = 0.f;
    for (int k = 0; k < SD; ++k) g += state[b * SD + k] * w_sgate[k * SD + j];
    for (int k = 0; k < SD; ++k) g += ni[b][k] * w_sgate[(SD + k) * SD + j];
    g = 1.f / (1.f + __expf(-g));
    outState[b * SD + j] = state[b * SD + j] * (1.f - g) + ni[b][j] * g;
}

// ---------------------------------------------------------------------------
extern "C" void kernel_launch(void* const* d_in, const int* in_sizes, int n_in,
                              void* d_out, int out_size, void* d_ws, size_t ws_size,
                              hipStream_t stream) {
    const int B = 2, S = 2048, D = 1024, H = 16, HD = 64, SD = 64;
    const float* x       = (const float*)d_in[0];
    const float* gamma   = (const float*)d_in[1];
    const float* beta    = (const float*)d_in[2];
    const float* state   = (const float*)d_in[3];
    const float* w_qkv   = (const float*)d_in[4];
    const float* w_o     = (const float*)d_in[5];
    const float* w_gate  = (const float*)d_in[6];
    const float* w_up    = (const float*)d_in[7];
    const float* w_down  = (const float*)d_in[8];
    const float* norm1_w = (const float*)d_in[9];
    const float* norm2_w = (const float*)d_in[10];
    const float* w_sm    = (const float*)d_in[11];
    const float* w_h2s   = (const float*)d_in[12];
    const float* w_sgate = (const float*)d_in[13];
    float* xout = (float*)d_out;                       // (B,S,D) then state (B,SD)

    char* ws = (char*)d_ws;
    size_t off = 0;
    auto arena = [&](size_t bytes) -> void* {
        void* p = ws + off;
        off = (off + bytes + 255) & ~(size_t)255;
        return p;
    };
    float* fg             = (float*)arena((size_t)B * D * 4);
    float* fb             = (float*)arena((size_t)B * D * 4);
    unsigned short* hbuf  = (unsigned short*)arena((size_t)B * S * D * 2);       // h / h2
    unsigned short* wqkvp = (unsigned short*)arena((size_t)D * 3 * D * 2);
    unsigned short* wop   = (unsigned short*)arena((size_t)D * D * 2);
    unsigned short* wgp   = (unsigned short*)arena((size_t)D * 3 * D * 2);
    unsigned short* wupp  = (unsigned short*)arena((size_t)D * 3 * D * 2);
    unsigned short* wdp   = (unsigned short*)arena((size_t)3 * D * D * 2);
    unsigned short* qkvb  = (unsigned short*)arena((size_t)B * S * 3 * D * 2);   // also act
    unsigned short* vp    = (unsigned short*)arena((size_t)B * H * (S / 32) * 4 * 512 * 2);
    unsigned short* attnb = (unsigned short*)arena((size_t)B * S * D * 2);
    float* x1             = (float*)arena((size_t)B * S * D * 4);
    unsigned short* gateb = (unsigned short*)arena((size_t)B * S * 3 * D * 2);
    float* meanb          = (float*)arena((size_t)B * D * 4);
    unsigned short* act   = qkvb;                      // reuse after attention

    const int MR = B * S;                              // 4096 rows

    // FiLM params
    film_kernel<<<(B * 2 * D + 255) / 256, 256, 0, stream>>>(state, w_sm, gamma, beta, fg, fb, SD, D);

    // pack weights -> B fragments (bf16)
    pack_b<<<(D * 3 * D + 255) / 256, 256, 0, stream>>>(w_qkv,  wqkvp, D, 3 * D);
    pack_b<<<(D * D + 255) / 256,     256, 0, stream>>>(w_o,    wop,   D, D);
    pack_b<<<(D * 3 * D + 255) / 256, 256, 0, stream>>>(w_gate, wgp,   D, 3 * D);
    pack_b<<<(D * 3 * D + 255) / 256, 256, 0, stream>>>(w_up,   wupp,  D, 3 * D);
    pack_b<<<(3 * D * D + 255) / 256, 256, 0, stream>>>(w_down, wdp,   3 * D, D);

    // h = rmsnorm(x)*fg + fb
    rmsnorm_film<<<MR, 256, 0, stream>>>(x, norm1_w, fg, fb, hbuf, S, D);

    // qkv = h @ w_qkv   (bf16 out)
    gemm_bf16<0><<<dim3(3 * D / 256, MR / 64), 256, 0, stream>>>(
        hbuf, wqkvp, nullptr, nullptr, nullptr, qkvb, MR, 3 * D, D);

    // pack V into B fragments
    {
        int total = B * H * (S / 32) * 4 * 512;
        pack_v<<<(total + 255) / 256, 256, 0, stream>>>(qkvb, vp, S, H, HD, total);
    }

    // attention
    flash_attn<<<(B * H * (S / 16)) / 8, 256, 0, stream>>>(qkvb, vp, attnb, S, H, HD);

    // x1 = x + attn @ w_o
    gemm_bf16<1><<<dim3(D / 256, MR / 64), 256, 0, stream>>>(
        attnb, wop, x, nullptr, x1, nullptr, MR, D, D);

    // h2 = rmsnorm(x1)*fg + fb
    rmsnorm_film<<<MR, 256, 0, stream>>>(x1, norm2_w, fg, fb, hbuf, S, D);

    // gate = silu(h2 @ w_gate)   (bf16)
    gemm_bf16<2><<<dim3(3 * D / 256, MR / 64), 256, 0, stream>>>(
        hbuf, wgp, nullptr, nullptr, nullptr, gateb, MR, 3 * D, D);

    // act = gate * (h2 @ w_up)   (bf16)
    gemm_bf16<3><<<dim3(3 * D / 256, MR / 64), 256, 0, stream>>>(
        hbuf, wupp, nullptr, gateb, nullptr, act, MR, 3 * D, D);

    // xout = x1 + act @ w_down
    gemm_bf16<1><<<dim3(D / 256, MR / 64), 256, 0, stream>>>(
        act, wdp, x1, nullptr, xout, nullptr, MR, D, 3 * D);

    // state update
    colmean<<<(B * D + 255) / 256, 256, 0, stream>>>(xout, meanb, B, S, D);
    state_update<<<1, 128, 0, stream>>>(state, meanb, w_h2s, w_sgate,
                                        xout + (size_t)B * S * D, D, SD);
}
